// InterpretableMultiHeadAttention_79104707658328
// MI455X (gfx1250) — matrix operand order repaired and checked by reading, not compiled, not run on hardware
//
#include <hip/hip_runtime.h>

#define B_   8
#define S_   2048
#define H_   4
#define HID_ 512
#define DK_  128

typedef __attribute__((ext_vector_type(16))) __bf16 v16bf;
typedef __attribute__((ext_vector_type(8)))  float  v8f;
typedef __attribute__((ext_vector_type(4)))  unsigned int u32x4;
typedef unsigned short u16;
typedef unsigned int   u32;

union V16u { u16 s[16]; u32x4 q[2]; v16bf v; };

__device__ __forceinline__ u16 f2bf(float f) {
  u32 u = __builtin_bit_cast(u32, f);
  u32 r = u + 0x7FFFu + ((u >> 16) & 1u);   // round-to-nearest-even
  return (u16)(r >> 16);
}

__device__ __forceinline__ v8f zero8() {
  v8f z = {0.f,0.f,0.f,0.f,0.f,0.f,0.f,0.f};
  return z;
}

__device__ __forceinline__ v8f wmma_bf16(v16bf a, v16bf b, v8f c) {
  return __builtin_amdgcn_wmma_f32_16x16x32_bf16(false, a, false, b, (short)0, c, false, false);
}

// A operand (16x32, 16-bit): lane m=L%16, hi=L/16; K runs [8*hi,8*hi+8) and [16+8*hi,...)
__device__ __forceinline__ v16bf load_a16(const u16* src, int ld, int lane) {
  const int m = lane & 15, hi = lane >> 4;
  const u16* p = src + m * ld + hi * 8;
  V16u t;
  t.q[0] = *(const u32x4*)(p);
  t.q[1] = *(const u32x4*)(p + 16);
  return t.v;
}

// B operand (32x16, 16-bit): lane n=L%16 (output col), hi=L/16; K runs [8*hi,8*hi+8) and [16+8*hi,...)
// src must be laid out so each output column's K values are contiguous (i.e. "transposed").
__device__ __forceinline__ v16bf load_b16(const u16* src, int ld, int lane) {
  const int nn = lane & 15, hi = lane >> 4;
  const u16* p = src + nn * ld + hi * 8;
  V16u t;
  t.q[0] = *(const u32x4*)(p);
  t.q[1] = *(const u32x4*)(p + 16);
  return t.v;
}

// ---------------- prep kernels ----------------
__global__ void k_prep_x(const float* __restrict__ x, u16* __restrict__ xbf) {
  const int n = B_ * S_ * HID_;
  for (int i = blockIdx.x * blockDim.x + threadIdx.x; i < n; i += gridDim.x * blockDim.x)
    xbf[i] = f2bf(x[i]);
}

__global__ void k_prep_w(const float* __restrict__ Wq, const float* __restrict__ Wk,
                         const float* __restrict__ Wv, const float* __restrict__ Wh,
                         u16* __restrict__ wqT, u16* __restrict__ wkT,
                         u16* __restrict__ wvT, u16* __restrict__ whT) {
  const int i = blockIdx.x * blockDim.x + threadIdx.x;
  if (i < H_ * DK_ * HID_) {          // [h][dk][hid] <- [h][hid][dk]
    const int hid = i % HID_;
    const int dk  = (i / HID_) % DK_;
    const int h   = i / (HID_ * DK_);
    const float scale = 0.08838834764831845f;  // 1/sqrt(128), folded into Q
    wqT[i] = f2bf(Wq[(h * HID_ + hid) * DK_ + dk] * scale);
    wkT[i] = f2bf(Wk[(h * HID_ + hid) * DK_ + dk]);
  }
  if (i < DK_ * HID_) {               // [dk][hid] <- [hid][dk]
    const int hid = i % HID_;
    const int dk  = i / HID_;
    wvT[i] = f2bf(Wv[hid * DK_ + dk]);
  }
  if (i < HID_ * DK_) {               // [hid][dk] <- [dk][hid]
    const int dk  = i % DK_;
    const int hid = i / DK_;
    whT[i] = f2bf(Wh[dk * HID_ + hid]);
  }
}

// ---------------- projection GEMM: x @ [Wv | Wq[h] | Wk[h]] + bias ----------------
// one wave = one 16-row s-tile x one 16-col tile of one of 9 128-col segments
__global__ void __launch_bounds__(256) k_proj(
    const u16* __restrict__ xbf,
    const u16* __restrict__ wvT, const u16* __restrict__ wqT, const u16* __restrict__ wkT,
    const float* __restrict__ bv, const float* __restrict__ bq, const float* __restrict__ bk,
    u16* __restrict__ qbuf, u16* __restrict__ kbuf, u16* __restrict__ vTt) {
  const int lane = threadIdx.x & 31;
  const int wave = (blockIdx.x * blockDim.x + threadIdx.x) >> 5;
  const int ct = wave % 72;          // 72 = 9 segments * 8 col-tiles
  const int st = wave / 72;          // 1024 s-tiles
  if (st >= (B_ * S_) / 16) return;
  const int seg = ct >> 3;           // 0=v, 1..4=q heads, 5..8=k heads
  const int c16 = ct & 7;
  const int nn = lane & 15, hi = lane >> 4;
  const int bidx = st >> 7;                  // S/16 = 128 tiles per batch
  const int srow0 = (st & 127) << 4;

  const u16* xr = xbf + (size_t)st * 16 * HID_;
  const u16* wt;
  const float* bias;
  float bscale = 1.0f;
  int h = 0;
  if (seg == 0)      { wt = wvT + (size_t)(c16 * 16) * HID_;              bias = bv + c16 * 16; }
  else if (seg <= 4) { h = seg - 1; wt = wqT + (size_t)(h * DK_ + c16 * 16) * HID_;
                       bias = bq + h * DK_ + c16 * 16; bscale = 0.08838834764831845f; }
  else               { h = seg - 5; wt = wkT + (size_t)(h * DK_ + c16 * 16) * HID_;
                       bias = bk + h * DK_ + c16 * 16; }

  v8f acc = zero8();
  #pragma unroll
  for (int d = 0; d < 16; ++d) {     // K = 512 in 16 steps of 32
    v16bf a = load_a16(xr + d * 32, HID_, lane);
    v16bf b = load_b16(wt + d * 32, HID_, lane);
    acc = wmma_bf16(a, b, acc);
  }
  const float bn = bias[nn] * bscale;

  if (seg == 0) {
    #pragma unroll
    for (int r = 0; r < 8; ++r) {
      const int s = srow0 + r + 8 * hi;
      vTt[((size_t)(bidx * DK_ + c16 * 16 + nn)) * S_ + s] = f2bf(acc[r] + bn);
    }
  } else if (seg <= 4) {
    #pragma unroll
    for (int r = 0; r < 8; ++r) {
      const int s = srow0 + r + 8 * hi;
      qbuf[(((size_t)(bidx * H_ + h)) * S_ + s) * DK_ + c16 * 16 + nn] = f2bf(acc[r] + bn);
    }
  } else {
    #pragma unroll
    for (int r = 0; r < 8; ++r) {
      const int s = srow0 + r + 8 * hi;
      kbuf[(((size_t)(bidx * H_ + h)) * S_ + s) * DK_ + c16 * 16 + nn] = f2bf(acc[r] + bn);
    }
  }
}

// ---------------- causal flash attention + attn materialization ----------------
// one wave = 16 query rows of one (b,h)
__global__ void __launch_bounds__(256) k_attn(
    const u16* __restrict__ qbuf, const u16* __restrict__ kbuf, const u16* __restrict__ vT,
    float* __restrict__ attn, float* __restrict__ head) {
  __shared__ alignas(16) u16 plds[8][16][32];   // per-wave P tile (D-layout -> A-layout bounce)
  const int lane = threadIdx.x & 31;
  const int wib  = threadIdx.x >> 5;
  const int wave = blockIdx.x * 8 + wib;
  const int stile = wave & 127;
  const int bh = wave >> 7;
  const int h = bh & (H_ - 1);
  const int bidx = bh >> 2;
  const int s0 = stile << 4;
  const int nn = lane & 15, hi = lane >> 4;

  const u16* qb = qbuf + (size_t)((bidx * H_ + h) * S_) * DK_;
  const u16* kb = kbuf + (size_t)((bidx * H_ + h) * S_) * DK_;
  const u16* vb = vT + (size_t)bidx * DK_ * S_;

  v16bf qa[4];                       // Q tile 16x128 resident (scale pre-folded)
  #pragma unroll
  for (int d = 0; d < 4; ++d) qa[d] = load_a16(qb + s0 * DK_ + d * 32, DK_, lane);

  float m_i[8], l_i[8];
  v8f O[8];
  #pragma unroll
  for (int r = 0; r < 8; ++r) { m_i[r] = -3.0e38f; l_i[r] = 0.f; }
  #pragma unroll
  for (int t = 0; t < 8; ++t) O[t] = zero8();

  const int nsteps = (s0 + 15) / 32 + 1;
  const float L2E = 1.4426950408889634f;

  // ---- sweep A: online softmax + PV accumulation ----
  for (int j = 0; j < nsteps; ++j) {
    const int t0 = j << 5;
    if (j + 1 < nsteps) __builtin_prefetch(kb + (size_t)(t0 + 32) * DK_, 0, 3);

    v8f c0 = zero8(), c1 = zero8();
    #pragma unroll
    for (int d = 0; d < 4; ++d) {
      v16bf b0 = load_b16(kb + (size_t)t0 * DK_ + d * 32, DK_, lane);
      v16bf b1 = load_b16(kb + (size_t)(t0 + 16) * DK_ + d * 32, DK_, lane);
      c0 = wmma_bf16(qa[d], b0, c0);
      c1 = wmma_bf16(qa[d], b1, c1);
    }

    float alpha[8];
    #pragma unroll
    for (int r = 0; r < 8; ++r) {
      const int sg = s0 + r + 8 * hi;
      if (t0 + nn > sg)      c0[r] = -3.0e38f;
      if (t0 + 16 + nn > sg) c1[r] = -3.0e38f;
      float mt = fmaxf(c0[r], c1[r]);
      mt = fmaxf(mt, __shfl_xor(mt, 1, 32));
      mt = fmaxf(mt, __shfl_xor(mt, 2, 32));
      mt = fmaxf(mt, __shfl_xor(mt, 4, 32));
      mt = fmaxf(mt, __shfl_xor(mt, 8, 32));
      const float mnew = fmaxf(m_i[r], mt);
      alpha[r] = __builtin_exp2f((m_i[r] - mnew) * L2E);
      m_i[r] = mnew;
      const float p0 = __builtin_exp2f((c0[r] - mnew) * L2E);
      const float p1 = __builtin_exp2f((c1[r] - mnew) * L2E);
      float ps = p0 + p1;
      ps += __shfl_xor(ps, 1, 32);
      ps += __shfl_xor(ps, 2, 32);
      ps += __shfl_xor(ps, 4, 32);
      ps += __shfl_xor(ps, 8, 32);
      l_i[r] = l_i[r] * alpha[r] + ps;
      plds[wib][r + 8 * hi][nn]      = f2bf(p0);   // D-layout -> row-major LDS
      plds[wib][r + 8 * hi][16 + nn] = f2bf(p1);
    }

    #pragma unroll
    for (int dt = 0; dt < 8; ++dt)
      #pragma unroll
      for (int r = 0; r < 8; ++r)
        O[dt][r] *= alpha[r];

    const v16bf pa = load_a16(&plds[wib][0][0], 32, lane);  // A-layout P (16x32)
    #pragma unroll
    for (int dt = 0; dt < 8; ++dt) {
      const v16bf bvv = load_b16(vb + (size_t)(dt * 16) * S_ + t0, S_, lane);
      O[dt] = wmma_bf16(pa, bvv, O[dt]);
    }
  }

  float inv_l[8];
  #pragma unroll
  for (int r = 0; r < 8; ++r) inv_l[r] = 1.0f / l_i[r];

  {   // head = O / l   (fp32 workspace, consumed by k_out)
    const size_t rowbase = (size_t)((bidx * H_ + h) * S_ + s0);
    #pragma unroll
    for (int dt = 0; dt < 8; ++dt)
      #pragma unroll
      for (int r = 0; r < 8; ++r)
        head[(rowbase + r + 8 * hi) * DK_ + dt * 16 + nn] = O[dt][r] * inv_l[r];
  }

  // ---- sweep B: recompute scores (L2-resident K), stream normalized attn ----
  for (int j = 0; j < nsteps; ++j) {
    const int t0 = j << 5;
    v8f c0 = zero8(), c1 = zero8();
    #pragma unroll
    for (int d = 0; d < 4; ++d) {
      v16bf b0 = load_b16(kb + (size_t)t0 * DK_ + d * 32, DK_, lane);
      v16bf b1 = load_b16(kb + (size_t)(t0 + 16) * DK_ + d * 32, DK_, lane);
      c0 = wmma_bf16(qa[d], b0, c0);
      c1 = wmma_bf16(qa[d], b1, c1);
    }
    #pragma unroll
    for (int r = 0; r < 8; ++r) {
      const int sg = s0 + r + 8 * hi;
      const float p0 = (t0 + nn > sg)      ? 0.f : __builtin_exp2f((c0[r] - m_i[r]) * L2E) * inv_l[r];
      const float p1 = (t0 + 16 + nn > sg) ? 0.f : __builtin_exp2f((c1[r] - m_i[r]) * L2E) * inv_l[r];
      float* ap = attn + (((size_t)(bidx * S_ + sg)) * H_ + h) * S_ + t0;
      ap[nn]      = p0;
      ap[16 + nn] = p1;
    }
  }

  // zero-fill strict upper-triangular remainder of each attn row
  const int tstart = nsteps << 5;
  for (int r = 0; r < 16; ++r) {
    float* ap = attn + (((size_t)(bidx * S_ + s0 + r)) * H_ + h) * S_;
    for (int t = tstart + lane; t < S_; t += 32) ap[t] = 0.f;
  }
}

// ---------------- out = mean_h(head) @ Wh ----------------
__global__ void __launch_bounds__(256) k_out(const float* __restrict__ head,
                                             const u16* __restrict__ whT,
                                             float* __restrict__ outp) {
  const int lane = threadIdx.x & 31;
  const int wave = (blockIdx.x * blockDim.x + threadIdx.x) >> 5;   // 0..1023 row tiles
  if (wave >= (B_ * S_) / 16) return;
  const int nn = lane & 15, hi = lane >> 4;
  const int rowflat = wave * 16 + nn;        // this lane's A-row
  const int bidx = rowflat >> 11;            // / 2048
  const int s = rowflat & 2047;

  v16bf a[4];                                // mean head 16x128, bf16, A-layout
  #pragma unroll
  for (int d = 0; d < 4; ++d) {
    float acc16[16];
    #pragma unroll
    for (int e = 0; e < 16; ++e) acc16[e] = 0.f;
    #pragma unroll
    for (int hh = 0; hh < H_; ++hh) {
      const float* p = head + (((size_t)(bidx * H_ + hh)) * S_ + s) * DK_ + d * 32 + hi * 8;
      #pragma unroll
      for (int e = 0; e < 8; ++e) acc16[e] += p[e];
      #pragma unroll
      for (int e = 0; e < 8; ++e) acc16[8 + e] += p[16 + e];
    }
    V16u t;
    #pragma unroll
    for (int e = 0; e < 16; ++e) t.s[e] = f2bf(acc16[e] * 0.25f);
    a[d] = t.v;
  }

  const size_t outrow0 = (size_t)wave * 16;
  for (int ctile = 0; ctile < 32; ++ctile) {   // 512 output cols
    v8f acc = zero8();
    #pragma unroll
    for (int d = 0; d < 4; ++d) {
      v16bf b = load_b16(whT + (size_t)(ctile * 16) * DK_ + d * 32, DK_, lane);
      acc = wmma_bf16(a[d], b, acc);
    }
    #pragma unroll
    for (int r = 0; r < 8; ++r)
      outp[(outrow0 + r + 8 * hi) * HID_ + ctile * 16 + nn] = acc[r];
  }
}

// ---------------- host ----------------
extern "C" void kernel_launch(void* const* d_in, const int* in_sizes, int n_in,
                              void* d_out, int out_size, void* d_ws, size_t ws_size,
                              hipStream_t stream) {
  const float* x  = (const float*)d_in[0];
  const float* Wv = (const float*)d_in[1];
  const float* bv = (const float*)d_in[2];
  const float* Wq = (const float*)d_in[3];
  const float* bq = (const float*)d_in[4];
  const float* Wk = (const float*)d_in[5];
  const float* bk = (const float*)d_in[6];
  const float* Wh = (const float*)d_in[7];
  float* outp = (float*)d_out;
  float* attn = outp + (size_t)B_ * S_ * HID_;

  char* ws = (char*)d_ws;
  size_t off = 0;
  auto alloc = [&](size_t bytes) {
    char* p = ws + off;
    off += (bytes + 255) & ~(size_t)255;
    return p;
  };
  u16*  xbf  = (u16*)alloc((size_t)B_ * S_ * HID_ * 2);
  u16*  qb   = (u16*)alloc((size_t)B_ * H_ * S_ * DK_ * 2);
  u16*  kb   = (u16*)alloc((size_t)B_ * H_ * S_ * DK_ * 2);
  u16*  vT   = (u16*)alloc((size_t)B_ * DK_ * S_ * 2);
  u16*  wqT  = (u16*)alloc((size_t)H_ * DK_ * HID_ * 2);
  u16*  wkT  = (u16*)alloc((size_t)H_ * DK_ * HID_ * 2);
  u16*  wvT  = (u16*)alloc((size_t)DK_ * HID_ * 2);
  u16*  whT  = (u16*)alloc((size_t)HID_ * DK_ * 2);
  float* head = (float*)alloc((size_t)B_ * H_ * S_ * DK_ * 4);

  k_prep_x<<<4096, 256, 0, stream>>>(x, xbf);
  k_prep_w<<<1024, 256, 0, stream>>>(Wq, Wk, Wv, Wh, wqT, wkT, wvT, whT);
  k_proj<<<9216, 256, 0, stream>>>(xbf, wvT, wqT, wkT, bv, bq, bk, qb, kb, vT);
  k_attn<<<512, 256, 0, stream>>>(qb, kb, vT, attn, head);
  k_out<<<128, 256, 0, stream>>>(head, whT, outp);
}
